// WanSelfAttention_57535381897884
// MI455X (gfx1250) — compile-verified, matrix operand order checked by reading, not run on hardware
//
#include <hip/hip_runtime.h>
#include <hip/hip_bf16.h>

// ---------------------------------------------------------------------------
// WanSelfAttention forward for gfx1250 (MI455X), wave32 + WMMA f16->f32,
// with TDM (tensor_load_to_lds) double-buffered staging of B/K/V tiles.
// ---------------------------------------------------------------------------

typedef __attribute__((ext_vector_type(16))) _Float16 v16h;
typedef __attribute__((ext_vector_type(8)))  _Float16 v8h;
typedef __attribute__((ext_vector_type(8)))  float    v8f;
typedef __attribute__((ext_vector_type(4)))  unsigned int u32x4;
typedef __attribute__((ext_vector_type(8)))  int      i32x8;
typedef __attribute__((ext_vector_type(4)))  int      i32x4;

union V16 { v16h v; v8h h2[2]; _Float16 e[16]; };
union V8F { v8f v; float f[8]; };

#define SEQ_LEN 3840          // 8*20*24
#define MODEL_D 1536
#define N_HEADS 12
#define HEAD_D  128
#define GRID_F  8
#define GRID_H  20
#define GRID_W  24

// --------------------------- TDM support -----------------------------------
#if defined(__gfx1250__) && __has_builtin(__builtin_amdgcn_tensor_load_to_lds)
#  define HAVE_TDM 1
#  if __has_include(<hip/amd_detail/amd_gfx1250_TDM.h>)
#    define TDM_SIX_ARG 1      // amdgpu-toolchain / therock-10.0 headers
#  else
#    define TDM_SIX_ARG 0      // ROCm 7.2 5-arg builtin
#  endif
#else
#  define HAVE_TDM 0
#endif

#if HAVE_TDM && __has_builtin(__builtin_amdgcn_s_wait_tensorcnt)
#  define WAIT_TENSOR() __builtin_amdgcn_s_wait_tensorcnt(0)
#else
#  define WAIT_TENSOR() ((void)0)
#endif

__device__ __forceinline__ unsigned lds_off(const void* p) {
    // Generic LDS pointer: addr[31:0] is the workgroup-relative LDS byte offset.
    return (unsigned)(unsigned long long)p;
}

// 1-D DMA: copy n8 * 8 bytes from global `ga` to LDS offset `loff`.
__device__ __forceinline__ void tdm_load_1d(unsigned loff, unsigned long long ga,
                                            unsigned n8) {
#if HAVE_TDM
    u32x4 g0 = { 1u,                                   // count = 1 valid descriptor
                 loff,                                 // lds_addr (bytes)
                 (unsigned)ga,                         // global_addr[31:0]
                 ((unsigned)((ga >> 32) & 0x01FFFFFFull)) | (2u << 30) }; // addr[56:32] | type=2
    i32x8 g1 = { (int)(3u << 16),                      // data_size = 3 (8 bytes)
                 (int)((n8 & 0xFFFFu) << 16),          // tensor_dim0[15:0]
                 (int)((n8 >> 16) | (1u << 16)),       // tensor_dim0[31:16] | tensor_dim1=1
                 (int)((n8 & 0xFFFFu) << 16),          // tile_dim0 = n8 (1-D tile)
                 0,                                    // tile_dim1 = tile_dim2 = 0 (unused)
                 (int)n8,                              // tensor_dim0_stride lo
                 0, 0 };
    i32x4 z4 = {0, 0, 0, 0};
#if TDM_SIX_ARG
    i32x8 z8 = {0, 0, 0, 0, 0, 0, 0, 0};
    __builtin_amdgcn_tensor_load_to_lds(g0, g1, z4, z4, z8, 0);
#else
    __builtin_amdgcn_tensor_load_to_lds(g0, g1, z4, z4, 0);
#endif
#else
    (void)loff; (void)ga; (void)n8;
#endif
}

// Synchronous cooperative fallback (all 256 threads), 16B per thread per step.
__device__ __forceinline__ void stage_copy(_Float16* dst, const _Float16* src,
                                           int halfs) {
    for (int i = threadIdx.x * 8; i < halfs; i += 256 * 8)
        *(v8h*)(dst + i) = *(const v8h*)(src + i);
}

// --------------------------- WMMA helpers ----------------------------------
__device__ __forceinline__ int kmap(int h, int j) {
    return (j < 8) ? (h * 8 + j) : (16 + h * 8 + (j - 8));
}

__device__ __forceinline__ v16h load_a16(const _Float16* p) {
    V16 u;
    u.h2[0] = *(const v8h*)(p);
    u.h2[1] = *(const v8h*)(p + 16);
    return u.v;
}

__device__ __forceinline__ v8f wmma_f16(v16h a, v16h b, v8f c) {
    return __builtin_amdgcn_wmma_f32_16x16x32_f16(false, a, false, b, (short)0, c,
                                                  false, false);
}

// ---------------------------------------------------------------------------
// Pack the 4 projection weights (fp32, K-major DIMxDIM) into f16 B-fragment
// tiles: per (kt, nt) 32x16 tile, lane-major (512 halfs contiguous).
// ---------------------------------------------------------------------------
__global__ void pack_weights_k(const float* __restrict__ qw, const float* __restrict__ kw,
                               const float* __restrict__ vw, const float* __restrict__ ow,
                               _Float16* __restrict__ dst) {
    const size_t per   = (size_t)MODEL_D * MODEL_D;
    const size_t total = 4 * per;
    for (size_t e = (size_t)blockIdx.x * blockDim.x + threadIdx.x; e < total;
         e += (size_t)gridDim.x * blockDim.x) {
        size_t w = e / per, r = e % per;
        size_t tile = r >> 9;
        int lidx = (int)(r & 511);
        int kt = (int)(tile / 96), nt = (int)(tile % 96);
        int lane = lidx >> 4, j = lidx & 15;
        int h = lane >> 4;
        int n = nt * 16 + (lane & 15);
        int k = kt * 32 + kmap(h, j);
        const float* W = (w == 0) ? qw : (w == 1) ? kw : (w == 2) ? vw : ow;
        dst[e] = (_Float16)W[(size_t)k * MODEL_D + n];
    }
}

__global__ void conv_half_k(const float* __restrict__ src, _Float16* __restrict__ dst,
                            size_t n) {
    for (size_t i = (size_t)blockIdx.x * blockDim.x + threadIdx.x; i < n;
         i += (size_t)gridDim.x * blockDim.x)
        dst[i] = (_Float16)src[i];
}

// ---------------------------------------------------------------------------
// GEMM: C(SEQ x 1536) = A_f16(SEQ x 1536) @ W_packed + bias   (fp32 out)
// Block = 8 waves stacked on M over one 64-col strip; B strip (4 KB per k-step)
// staged in LDS via TDM double-buffering and reused by all 8 waves.
// ---------------------------------------------------------------------------
__global__ void __launch_bounds__(256) gemm_k(const _Float16* __restrict__ A,
                                              const _Float16* __restrict__ Bp,
                                              const float* __restrict__ bias,
                                              float* __restrict__ C) {
    __shared__ __align__(32) _Float16 bstage[2][2048];   // 2 x 4 KB (4 B-tiles)
    int lane = threadIdx.x & 31;
    int wid  = threadIdx.x >> 5;
    int mtB  = blockIdx.x / 24;          // 0..29
    int nq   = blockIdx.x % 24;          // 64-col strip
    int mt   = mtB * 8 + wid;            // 0..239
    int h = lane >> 4, nn = lane & 15;

    const _Float16* bcol = Bp + ((size_t)(nq * 4) << 9);   // strip(kt) = bcol + kt*96*512

#if HAVE_TDM
    if (wid == 0) {
        tdm_load_1d(lds_off(&bstage[0][0]), (unsigned long long)bcol, 512);
        WAIT_TENSOR();
    }
#else
    stage_copy(&bstage[0][0], bcol, 2048);
#endif
    __syncthreads();

    v8f z = {};
    v8f acc0 = z, acc1 = z, acc2 = z, acc3 = z;
    const _Float16* arow = A + (size_t)(mt * 16 + nn) * MODEL_D;

    for (int kt = 0; kt < 48; ++kt) {
        int p = kt & 1;
        const _Float16* nextB = bcol + (size_t)(kt + 1) * 96 * 512;
#if HAVE_TDM
        if (wid == 0 && kt + 1 < 48)
            tdm_load_1d(lds_off(&bstage[1 - p][0]), (unsigned long long)nextB, 512);
#else
        if (kt + 1 < 48) stage_copy(&bstage[1 - p][0], nextB, 2048);
#endif
        if (kt + 1 < 48)
            __builtin_prefetch(arow + (kt + 1) * 32 + h * 8, 0, 0);

        v16h a = load_a16(arow + kt * 32 + h * 8);
        const v16h* bt = (const v16h*)&bstage[p][0];
        acc0 = wmma_f16(a, bt[0 * 32 + lane], acc0);
        acc1 = wmma_f16(a, bt[1 * 32 + lane], acc1);
        acc2 = wmma_f16(a, bt[2 * 32 + lane], acc2);
        acc3 = wmma_f16(a, bt[3 * 32 + lane], acc3);

#if HAVE_TDM
        if (wid == 0 && kt + 1 < 48) WAIT_TENSOR();
#endif
        __syncthreads();
    }

    v8f accs[4] = {acc0, acc1, acc2, acc3};
#pragma unroll
    for (int i = 0; i < 4; ++i) {
        V8F u; u.v = accs[i];
        int col = nq * 64 + i * 16 + nn;
        float bb = bias[col];
#pragma unroll
        for (int r = 0; r < 8; ++r) {
            int row = mt * 16 + r + 8 * h;
            C[(size_t)row * MODEL_D + col] = u.f[r] + bb;
        }
    }
}

// ---------------------------------------------------------------------------
// Fused RMSNorm + RoPE per sequence row; writes f16 row-major (A-frag ready).
// ---------------------------------------------------------------------------
__global__ void __launch_bounds__(256) rmsrope_k(const float* __restrict__ lin,
                                                 const float* __restrict__ nw,
                                                 const float* __restrict__ freqs,
                                                 _Float16* __restrict__ out) {
    __shared__ float red[256];
    __shared__ float s_inv;
    int s = blockIdx.x, tid = threadIdx.x;
    const float* row = lin + (size_t)s * MODEL_D;

    float acc = 0.f;
    for (int i = tid; i < MODEL_D; i += 256) { float v = row[i]; acc += v * v; }
    red[tid] = acc; __syncthreads();
    for (int st = 128; st > 0; st >>= 1) {
        if (tid < st) red[tid] += red[tid + st];
        __syncthreads();
    }
    if (tid == 0) s_inv = rsqrtf(red[0] / (float)MODEL_D + 1e-6f);
    __syncthreads();
    float inv = s_inv;

    int fpos = s / (GRID_H * GRID_W);
    int hpos = (s / GRID_W) % GRID_H;
    int wpos = s % GRID_W;

    for (int p = tid; p < MODEL_D / 2; p += 256) {
        int i = p & 63;                                      // pair idx in head (c=64)
        int pos = (i < 22) ? fpos : (i < 43) ? hpos : wpos;  // d_f=22, d_hw=21
        float ang = freqs[pos * 64 + i];
        float sn, cs;
        __sincosf(ang, &sn, &cs);
        float xr = row[2 * p]     * inv * nw[2 * p];
        float xi = row[2 * p + 1] * inv * nw[2 * p + 1];
        out[(size_t)s * MODEL_D + 2 * p]     = (_Float16)(xr * cs - xi * sn);
        out[(size_t)s * MODEL_D + 2 * p + 1] = (_Float16)(xr * sn + xi * cs);
    }
}

// ---------------------------------------------------------------------------
// Pack K (f16 row-major) into B fragments for QK^T.
// kB[((hd*240 + nt)*4 + kc)*512 + lane*16 + j]
// ---------------------------------------------------------------------------
__global__ void pack_kB_k(const _Float16* __restrict__ kh, _Float16* __restrict__ dst) {
    const size_t total = (size_t)N_HEADS * 240 * 4 * 512;
    for (size_t e = (size_t)blockIdx.x * blockDim.x + threadIdx.x; e < total;
         e += (size_t)gridDim.x * blockDim.x) {
        int lidx = (int)(e & 511);
        size_t t = e >> 9;
        int kc = (int)(t & 3);
        size_t t2 = t >> 2;
        int nt = (int)(t2 % 240);
        int hd = (int)(t2 / 240);
        int lane = lidx >> 4, j = lidx & 15;
        int h = lane >> 4;
        int n = nt * 16 + (lane & 15);
        int dim = hd * HEAD_D + kc * 32 + kmap(h, j);
        dst[e] = kh[(size_t)n * MODEL_D + dim];
    }
}

// Pack V (fp32 row-major) into B fragments for P@V.
// vB[((hd*120 + kt)*8 + ct)*512 + lane*16 + j]
__global__ void pack_vB_k(const float* __restrict__ vlin, _Float16* __restrict__ dst) {
    const size_t total = (size_t)N_HEADS * 120 * 8 * 512;
    for (size_t e = (size_t)blockIdx.x * blockDim.x + threadIdx.x; e < total;
         e += (size_t)gridDim.x * blockDim.x) {
        int lidx = (int)(e & 511);
        size_t t = e >> 9;
        int ct = (int)(t & 7);
        size_t t2 = t >> 3;
        int kt = (int)(t2 % 120);
        int hd = (int)(t2 / 120);
        int lane = lidx >> 4, j = lidx & 15;
        int h = lane >> 4;
        int key = kt * 32 + kmap(h, j);
        int col = hd * HEAD_D + ct * 16 + (lane & 15);
        dst[e] = (_Float16)vlin[(size_t)key * MODEL_D + col];
    }
}

// ---------------------------------------------------------------------------
// Flash attention: block = 8 waves, all in the SAME head (240 % 8 == 0), each
// wave owns one 16-query tile. Shared K/V fragment strips (8 KB each per
// 32-key step) staged in LDS via TDM double-buffering. Online softmax fp32.
// ---------------------------------------------------------------------------
__global__ void __launch_bounds__(256) flash_k(const _Float16* __restrict__ Q,
                                               const _Float16* __restrict__ Kb,
                                               const _Float16* __restrict__ Vb,
                                               _Float16* __restrict__ O) {
    __shared__ __align__(32) _Float16 kstage[2][4096];   // 2 x 8 KB
    __shared__ __align__(32) _Float16 vstage[2][4096];   // 2 x 8 KB
    __shared__ __align__(32) _Float16 plds[8][512];      // 16x32 P per wave
    int lane = threadIdx.x & 31, wid = threadIdx.x >> 5;
    int wv = blockIdx.x * 8 + wid;            // 0..2879
    int hd = wv / 240, qt = wv % 240;
    int h = lane >> 4, nn = lane & 15;

    // Per-block streams (same head for all 8 waves):
    const _Float16* kbase = Kb + ((size_t)(hd * 240 * 4) << 9);  // + nt*4*512
    const _Float16* vbase = Vb + ((size_t)(hd * 120 * 8) << 9);  // + kt*8*512

    v16h qa[4];
#pragma unroll
    for (int kc = 0; kc < 4; ++kc)
        qa[kc] = load_a16(Q + (size_t)(qt * 16 + nn) * MODEL_D + hd * HEAD_D + kc * 32 + h * 8);

    float mrow[8], lrow[8];
#pragma unroll
    for (int r = 0; r < 8; ++r) { mrow[r] = -1e30f; lrow[r] = 0.f; }
    v8f z = {};
    V8F o[8];
#pragma unroll
    for (int c = 0; c < 8; ++c) o[c].v = z;

    const float scale = 0.0883883476483184f;   // 1/sqrt(128)

#if HAVE_TDM
    if (wid == 0) {
        tdm_load_1d(lds_off(&kstage[0][0]), (unsigned long long)kbase, 1024);
        tdm_load_1d(lds_off(&vstage[0][0]), (unsigned long long)vbase, 1024);
        WAIT_TENSOR();
    }
#else
    stage_copy(&kstage[0][0], kbase, 4096);
    stage_copy(&vstage[0][0], vbase, 4096);
#endif
    __syncthreads();

    for (int kt = 0; kt < 120; ++kt) {
        int p = kt & 1;
#if HAVE_TDM
        if (wid == 0 && kt + 1 < 120) {
            tdm_load_1d(lds_off(&kstage[1 - p][0]),
                        (unsigned long long)(kbase + (size_t)(2 * (kt + 1)) * 4 * 512), 1024);
            tdm_load_1d(lds_off(&vstage[1 - p][0]),
                        (unsigned long long)(vbase + (size_t)(kt + 1) * 8 * 512), 1024);
        }
#else
        if (kt + 1 < 120) {
            stage_copy(&kstage[1 - p][0], kbase + (size_t)(2 * (kt + 1)) * 4 * 512, 4096);
            stage_copy(&vstage[1 - p][0], vbase + (size_t)(kt + 1) * 8 * 512, 4096);
        }
#endif
        v8f s0v = z, s1v = z;
        const v16h* kb0 = (const v16h*)&kstage[p][0];
        const v16h* kb1 = kb0 + 128;               // second 16-key tile group
#pragma unroll
        for (int kc = 0; kc < 4; ++kc) {
            s0v = wmma_f16(qa[kc], kb0[kc * 32 + lane], s0v);
            s1v = wmma_f16(qa[kc], kb1[kc * 32 + lane], s1v);
        }
        V8F s0, s1; s0.v = s0v; s1.v = s1v;

#pragma unroll
        for (int r = 0; r < 8; ++r) {
            float v0 = s0.f[r] * scale, v1 = s1.f[r] * scale;
            float mx = fmaxf(v0, v1);
#pragma unroll
            for (int off = 1; off < 16; off <<= 1)
                mx = fmaxf(mx, __shfl_xor(mx, off, 32));    // reduce within 16-lane half
            float mnew = fmaxf(mrow[r], mx);
            float p0 = __expf(v0 - mnew), p1 = __expf(v1 - mnew);
            float ps = p0 + p1;
#pragma unroll
            for (int off = 1; off < 16; off <<= 1)
                ps += __shfl_xor(ps, off, 32);
            float alpha = __expf(mrow[r] - mnew);
            lrow[r] = lrow[r] * alpha + ps;
            mrow[r] = mnew;
#pragma unroll
            for (int c = 0; c < 8; ++c) o[c].f[r] *= alpha;
            plds[wid][(r + 8 * h) * 32 + nn]      = (_Float16)p0;
            plds[wid][(r + 8 * h) * 32 + 16 + nn] = (_Float16)p1;
        }
        __syncthreads();

        v16h pa = load_a16(&plds[wid][nn * 32 + h * 8]);    // A-frag of P
        const v16h* vb = (const v16h*)&vstage[p][0];
#pragma unroll
        for (int c = 0; c < 8; ++c)
            o[c].v = wmma_f16(pa, vb[c * 32 + lane], o[c].v);

#if HAVE_TDM
        if (wid == 0 && kt + 1 < 120) WAIT_TENSOR();
#endif
        __syncthreads();
    }

#pragma unroll
    for (int c = 0; c < 8; ++c) {
        int col = hd * HEAD_D + c * 16 + nn;
#pragma unroll
        for (int r = 0; r < 8; ++r) {
            int row = qt * 16 + r + 8 * h;
            O[(size_t)row * MODEL_D + col] = (_Float16)(o[c].f[r] / lrow[r]);
        }
    }
}

// ---------------------------------------------------------------------------
extern "C" void kernel_launch(void* const* d_in, const int* in_sizes, int n_in,
                              void* d_out, int out_size, void* d_ws, size_t ws_size,
                              hipStream_t stream) {
    (void)in_sizes; (void)n_in; (void)out_size; (void)ws_size;
    const float* x   = (const float*)d_in[0];
    const float* qw  = (const float*)d_in[1];
    const float* qb  = (const float*)d_in[2];
    const float* kw  = (const float*)d_in[3];
    const float* kb  = (const float*)d_in[4];
    const float* vw  = (const float*)d_in[5];
    const float* vb  = (const float*)d_in[6];
    const float* ow  = (const float*)d_in[7];
    const float* ob  = (const float*)d_in[8];
    const float* nqw = (const float*)d_in[9];
    const float* nkw = (const float*)d_in[10];
    const float* fr  = (const float*)d_in[11];
    // d_in[12] = grid_sizes (8,20,24) — fixed, baked in.

    char* ws = (char*)d_ws;
    size_t off = 0;
    auto carve = [&](size_t bytes) -> char* {
        char* p = ws + off;
        off += (bytes + 255) & ~(size_t)255;
        return p;
    };
    const size_t SD2 = (size_t)SEQ_LEN * MODEL_D * 2;   // f16 S x D
    const size_t SD4 = (size_t)SEQ_LEN * MODEL_D * 4;   // f32 S x D
    _Float16* pw   = (_Float16*)carve(4ull * MODEL_D * MODEL_D * 2);
    _Float16* xh   = (_Float16*)carve(SD2);
    float*    qlin = (float*)carve(SD4);
    float*    klin = (float*)carve(SD4);
    float*    vlin = (float*)carve(SD4);
    _Float16* qh   = (_Float16*)carve(SD2);
    _Float16* kh   = (_Float16*)carve(SD2);
    _Float16* kB   = (_Float16*)carve(SD2);
    _Float16* vB   = (_Float16*)carve(SD2);
    _Float16* attn = (_Float16*)qlin;   // qlin dead after rmsrope; reuse for attn out

    const size_t WW = (size_t)MODEL_D * MODEL_D;        // halfs per packed weight

    pack_weights_k<<<2048, 256, 0, stream>>>(qw, kw, vw, ow, pw);
    conv_half_k<<<2048, 256, 0, stream>>>(x, xh, (size_t)SEQ_LEN * MODEL_D);

    gemm_k<<<720, 256, 0, stream>>>(xh, pw + 0 * WW, qb, qlin);
    gemm_k<<<720, 256, 0, stream>>>(xh, pw + 1 * WW, kb, klin);
    gemm_k<<<720, 256, 0, stream>>>(xh, pw + 2 * WW, vb, vlin);

    rmsrope_k<<<SEQ_LEN, 256, 0, stream>>>(qlin, nqw, fr, qh);
    rmsrope_k<<<SEQ_LEN, 256, 0, stream>>>(klin, nkw, fr, kh);

    pack_kB_k<<<2048, 256, 0, stream>>>(kh, kB);
    pack_vB_k<<<2048, 256, 0, stream>>>(vlin, vB);

    flash_k<<<360, 256, 0, stream>>>(qh, kB, vB, attn);

    gemm_k<<<720, 256, 0, stream>>>(attn, pw + 3 * WW, ob, (float*)d_out);
}
